// GRUCell_32435593019450
// MI455X (gfx1250) — matrix operand order repaired and checked by reading, not compiled, not run on hardware
//
#include <hip/hip_runtime.h>
#include <hip/hip_bf16.h>

// ---------------------------------------------------------------------------
// GRU (B=64, S=512, I=512, H=512) for gfx1250 / MI455X.
// Phase 0: fp32 -> bf16 conversion of x, W (split x-part / h-part), h0.
// Phase 1: pre[g][t][b][j] = x @ Wg_x^T + b_g   (bf16 WMMA, f32 accum, f32 out)
// Phase 2: persistent scan kernel, 32 blocks, grid barrier per step,
//          register-resident recurrent weights (16 x v16bf per wave).
// ---------------------------------------------------------------------------

typedef __attribute__((ext_vector_type(16))) __bf16 v16bf;
typedef __attribute__((ext_vector_type(8)))  float  v8f;
typedef __attribute__((ext_vector_type(4)))  unsigned int u32x4;

union ABFrag { v16bf v; u32x4 q[2]; };

#define GB 64      // batch
#define GS 512     // seq len
#define GI 512     // input dim
#define GH 512     // hidden dim
#define GIH 1024   // I + H
#define NBLK_SCAN 32

__device__ __forceinline__ float sigmoidf_(float x) {
  return 1.0f / (1.0f + __expf(-x));
}

// ---------------- Phase 0: conversions ----------------
__global__ void k_cvt_x(const float* __restrict__ x, __bf16* __restrict__ xbf, long n) {
  long i = (long)blockIdx.x * blockDim.x + threadIdx.x;
  long stride = (long)gridDim.x * blockDim.x;
  for (; i < n; i += stride) xbf[i] = (__bf16)x[i];
}

__global__ void k_prep_w(const float* __restrict__ Wr, const float* __restrict__ Wz,
                         const float* __restrict__ Wh,
                         __bf16* __restrict__ wx, __bf16* __restrict__ whh) {
  long n = 3L * GH * GIH;
  long i = (long)blockIdx.x * blockDim.x + threadIdx.x;
  long stride = (long)gridDim.x * blockDim.x;
  for (; i < n; i += stride) {
    int g = (int)(i >> 19);            // GH*GIH = 2^19
    long rem = i & ((1L << 19) - 1);
    int nrow = (int)(rem >> 10);       // GIH = 2^10
    int c = (int)(rem & 1023);
    const float* W = (g == 0) ? Wr : ((g == 1) ? Wz : Wh);
    __bf16 v = (__bf16)W[(long)nrow * GIH + c];
    if (c < GI) wx[((long)g * GH + nrow) * GI + c] = v;
    else        whh[((long)g * GH + nrow) * GH + (c - GI)] = v;
  }
}

__global__ void k_init_h(const float* __restrict__ h, __bf16* __restrict__ hb) {
  int i = blockIdx.x * blockDim.x + threadIdx.x;
  if (i < GB * GH) hb[i] = (__bf16)h[i];
}

// ---------------- Phase 1: input projections ----------------
// One wave per 16x16 output tile.  M = B*S = 32768 (2048 tiles),
// N = 3*H = 1536 (96 tiles => g = matrix, j0 = column).  K = I = 512.
__global__ __launch_bounds__(256) void k_xproj(
    const __bf16* __restrict__ xbf, const __bf16* __restrict__ wxbf,
    const float* __restrict__ br, const float* __restrict__ bz,
    const float* __restrict__ bh,
    float* __restrict__ preR, float* __restrict__ preZ, float* __restrict__ preH) {
  int lane = threadIdx.x & 31;
  int wid  = threadIdx.x >> 5;
  int wave = blockIdx.x * 8 + wid;          // 0 .. 2048*96-1
  const int NT = 96;
  int mt = wave / NT;
  int nt = wave - mt * NT;
  int g  = nt >> 5;                          // 0..2 (r,z,h)
  int j0 = (nt & 31) * 16;
  int m0 = mt * 16;
  int ln = lane & 15;
  int hi = lane >> 4;

  // A: row m0+ln of x (bf16), chunks K {k0 + 8*hi .. +7} and {+16}
  const __bf16* arow = xbf + (size_t)(m0 + ln) * GI + hi * 8;
  // B: row (g*H + j0 + ln) of Wg_x, chunks K {k0 + 8*hi .. +7} and {+16}
  const __bf16* brow = wxbf + ((size_t)g * GH + j0 + ln) * GI + hi * 8;

  v8f acc = {0.f, 0.f, 0.f, 0.f, 0.f, 0.f, 0.f, 0.f};
#pragma unroll
  for (int kk = 0; kk < 16; ++kk) {
    int k0 = kk * 32;
    ABFrag a, b;
    a.q[0] = *(const u32x4*)(arow + k0);
    a.q[1] = *(const u32x4*)(arow + k0 + 16);
    b.q[0] = *(const u32x4*)(brow + k0);
    b.q[1] = *(const u32x4*)(brow + k0 + 16);
    acc = __builtin_amdgcn_wmma_f32_16x16x32_bf16(
        false, a.v, false, b.v, (short)0, acc, false, false);
  }

  int j = j0 + ln;
  const float* bias = (g == 0) ? br : ((g == 1) ? bz : bh);
  float bj = bias[j];
  float* pre = (g == 0) ? preR : ((g == 1) ? preZ : preH);
#pragma unroll
  for (int v = 0; v < 8; ++v) {
    int m  = m0 + v + 8 * hi;   // global row in B*S
    int bb = m >> 9;            // / S
    int ss = m & (GS - 1);
    pre[((size_t)ss * GB + bb) * GH + j] = acc[v] + bj;   // layout [t][b][j]
  }
}

// ---------------- Phase 2: recurrent scan (persistent) ----------------
// 32 blocks, each owns a 16-column strip j0 = 16*blockIdx.x of H.
// 384 threads = 12 waves: wave w -> g = w>>2 (matrix r/z/h), mt = w&3 (M tile).
// Recurrent B-fragments (16 x v16bf = 128 VGPRs) preloaded once.
__global__ __launch_bounds__(384, 1) void k_scan(
    const __bf16* __restrict__ whbf,
    const float* __restrict__ preR, const float* __restrict__ preZ,
    const float* __restrict__ preH,
    __bf16* hb0, __bf16* hb1,
    float* __restrict__ y, float* __restrict__ hlast, int* cnt) {
  __shared__ float s_r[4][16][16];
  __shared__ float s_z[4][16][16];

  int tid  = threadIdx.x;
  int lane = tid & 31;
  int wid  = tid >> 5;
  int g    = wid >> 2;       // 0:r 1:z 2:h~
  int mt   = wid & 3;        // M tile (batch rows mt*16..mt*16+15)
  int j0   = blockIdx.x * 16;
  int ln   = lane & 15;
  int hi   = lane >> 4;
  int m0   = mt * 16;
  int j    = j0 + ln;

  // Preload recurrent weight fragments: W_g_h rows j0+ln, all K=512.
  const __bf16* brow = whbf + ((size_t)g * GH + j0 + ln) * GH + hi * 8;
  ABFrag w[16];
#pragma unroll
  for (int kk = 0; kk < 16; ++kk) {
    w[kk].q[0] = *(const u32x4*)(brow + kk * 32);
    w[kk].q[1] = *(const u32x4*)(brow + kk * 32 + 16);
  }

  __bf16* hb[2] = {hb0, hb1};

  for (int t = 0; t < GS; ++t) {
    const __bf16* hc = hb[t & 1];
    __bf16*       hn = hb[(t + 1) & 1];

    // A fragments: rows m0+ln of current h (bf16).
    const __bf16* arow = hc + (size_t)(m0 + ln) * GH + hi * 8;
    v8f acc = {0.f, 0.f, 0.f, 0.f, 0.f, 0.f, 0.f, 0.f};
#pragma unroll
    for (int kk = 0; kk < 16; ++kk) {
      ABFrag a;
      a.q[0] = *(const u32x4*)(arow + kk * 32);
      a.q[1] = *(const u32x4*)(arow + kk * 32 + 16);
      acc = __builtin_amdgcn_wmma_f32_16x16x32_bf16(
          false, a.v, false, w[kk].v, (short)0, acc, false, false);
    }

    size_t preoff = (size_t)t * GB * GH;   // [t][b][j]
    if (g == 0) {
#pragma unroll
      for (int v = 0; v < 8; ++v) {
        int mr = v + 8 * hi;
        float xr = preR[preoff + (size_t)(m0 + mr) * GH + j];
        s_r[mt][mr][ln] = sigmoidf_(xr + acc[v]);
      }
      if (t + 1 < GS)
        __builtin_prefetch(preR + preoff + (size_t)GB * GH + (size_t)m0 * GH + j, 0, 1);
    } else if (g == 1) {
#pragma unroll
      for (int v = 0; v < 8; ++v) {
        int mr = v + 8 * hi;
        float xz = preZ[preoff + (size_t)(m0 + mr) * GH + j];
        s_z[mt][mr][ln] = sigmoidf_(xz + acc[v]);
      }
      if (t + 1 < GS)
        __builtin_prefetch(preZ + preoff + (size_t)GB * GH + (size_t)m0 * GH + j, 0, 1);
    }
    __syncthreads();
    if (g == 2) {
#pragma unroll
      for (int v = 0; v < 8; ++v) {
        int mr = v + 8 * hi;
        int m  = m0 + mr;                 // batch index b
        float r  = s_r[mt][mr][ln];
        float z  = s_z[mt][mr][ln];
        float xh = preH[preoff + (size_t)m * GH + j];
        float ho = (float)hc[(size_t)m * GH + j];
        float ht = tanhf(xh + r * acc[v]);
        float hnew = (1.0f - z) * ho + z * ht;
        y[((size_t)m * GS + t) * GH + j] = hnew;
        hn[(size_t)m * GH + j] = (__bf16)hnew;
        if (t == GS - 1) hlast[(size_t)m * GH + j] = hnew;
      }
      if (t + 1 < GS)
        __builtin_prefetch(preH + preoff + (size_t)GB * GH + (size_t)m0 * GH + j, 0, 1);
    }

    // ---- grid-wide barrier (monotonic counter) ----
    __threadfence();
    __syncthreads();
    if (tid == 0) {
      atomicAdd(cnt, 1);
      int target = NBLK_SCAN * (t + 1);
      while (__hip_atomic_load(cnt, __ATOMIC_ACQUIRE, __HIP_MEMORY_SCOPE_AGENT) < target) {
        __builtin_amdgcn_s_sleep(1);
      }
    }
    __syncthreads();
    __threadfence();
  }
}

// ---------------------------------------------------------------------------
extern "C" void kernel_launch(void* const* d_in, const int* in_sizes, int n_in,
                              void* d_out, int out_size, void* d_ws, size_t ws_size,
                              hipStream_t stream) {
  const float* x  = (const float*)d_in[0];
  const float* h0 = (const float*)d_in[1];
  const float* Wr = (const float*)d_in[2];
  const float* br = (const float*)d_in[3];
  const float* Wz = (const float*)d_in[4];
  const float* bz = (const float*)d_in[5];
  const float* Wh = (const float*)d_in[6];
  const float* bh = (const float*)d_in[7];

  float* y     = (float*)d_out;
  float* hlast = y + (size_t)GB * GS * GH;

  char* ws = (char*)d_ws;
  size_t off = 0;
  auto alloc = [&](size_t bytes) -> void* {
    off = (off + 255) & ~(size_t)255;
    void* p = ws + off;
    off += bytes;
    return p;
  };
  float*  preR = (float*)alloc((size_t)GS * GB * GH * sizeof(float));
  float*  preZ = (float*)alloc((size_t)GS * GB * GH * sizeof(float));
  float*  preH = (float*)alloc((size_t)GS * GB * GH * sizeof(float));
  __bf16* xbf  = (__bf16*)alloc((size_t)GB * GS * GI * 2);
  __bf16* wxbf = (__bf16*)alloc((size_t)3 * GH * GI * 2);
  __bf16* whbf = (__bf16*)alloc((size_t)3 * GH * GH * 2);
  __bf16* hbA  = (__bf16*)alloc((size_t)GB * GH * 2);
  __bf16* hbB  = (__bf16*)alloc((size_t)GB * GH * 2);
  int*    cnt  = (int*)alloc(256);

  hipMemsetAsync(cnt, 0, 256, stream);

  k_cvt_x <<<8192, 256, 0, stream>>>(x, xbf, (long)GB * GS * GI);
  k_prep_w<<<4096, 256, 0, stream>>>(Wr, Wz, Wh, wxbf, whbf);
  k_init_h<<<(GB * GH + 255) / 256, 256, 0, stream>>>(h0, hbA);

  // 2048 M-tiles * 96 N-tiles, 8 waves per block
  k_xproj<<<24576, 256, 0, stream>>>(xbf, wxbf, br, bz, bh, preR, preZ, preH);

  k_scan<<<NBLK_SCAN, 384, 0, stream>>>(whbf, preR, preZ, preH,
                                        hbA, hbB, y, hlast, cnt);
}
